// DyMixprop_24790551233337
// MI455X (gfx1250) — matrix operand order repaired and checked by reading, not compiled, not run on hardware
//
#include <hip/hip_runtime.h>

// ---------------- problem constants ----------------
#define ALPHA 0.05f
#define NB 32
#define CI 32
#define VN 400
#define LN 12
#define NT 25          // spatial tiles of 16 (400 = 25*16)
#define TS 40          // row stride (elems) for [*][32] bf16 tile buffers (bank-friendly, 16B aligned)
#define GS 424         // row stride for gA [32][424] bf16
#define WS 104         // row stride for Wm [32][96->104] bf16

typedef __attribute__((ext_vector_type(16))) __bf16 v16bf;
typedef __attribute__((ext_vector_type(8)))  float  v8f;
typedef __attribute__((ext_vector_type(4)))  unsigned int v4u;

union Frag { v16bf v; v4u q[2]; };
union Acc  { v8f  v; float f[8]; };

// ---------------- LDS layout (dynamic shared) ----------------
constexpr int SZ_T    = 416 * TS * 2;            // 33280
constexpr int OFF_XT  = 0;
constexpr int OFF_X1T = OFF_XT  + SZ_T;
constexpr int OFF_X2T = OFF_X1T + SZ_T;
constexpr int SZ_GA   = CI * GS * 2;             // 27136
constexpr int OFF_GA0 = OFF_X2T + SZ_T;
constexpr int OFF_GA1 = OFF_GA0 + SZ_GA;
constexpr int SZ_EB   = NT * 16 * TS * 2;        // 32000 (per-wave E / hT scratch, aliased)
constexpr int OFF_EB  = OFF_GA1 + SZ_GA;
constexpr int OFF_W1  = OFF_EB  + SZ_EB;
constexpr int OFF_W2  = OFF_W1  + 32 * TS * 2;
constexpr int OFF_WM1 = OFF_W2  + 32 * TS * 2;
constexpr int OFF_WM2 = OFF_WM1 + 32 * WS * 2;
constexpr int OFF_Z0  = OFF_WM2 + 32 * WS * 2;
constexpr int OFF_Z1  = OFF_Z0  + 416 * 4;
constexpr int OFF_BL1 = OFF_Z1  + 416 * 4;
constexpr int OFF_BL2 = OFF_BL1 + 128;
constexpr int OFF_BM  = OFF_BL2 + 128;
constexpr int SMEM_BYTES = OFF_BM + 128;         // 208256 B < 320KB/WGP

__device__ __forceinline__ unsigned short f2bf(float x) {
  unsigned int b = __float_as_uint(x);
  b += 0x7FFFu + ((b >> 16) & 1u);               // RNE
  return (unsigned short)(b >> 16);
}
__device__ __forceinline__ float bf2f(unsigned short h) {
  return __uint_as_float(((unsigned int)h) << 16);
}
// single-instruction hardware reciprocal (v_rcp_f32, TRANS)
__device__ __forceinline__ float fast_rcp(float x) { return __builtin_amdgcn_rcpf(x); }
// branchless saturating tanh: 1 - 2/(exp(2x)+1); exp->inf => 1, exp->0 => -1.
// lowers to v_exp_f32 + v_rcp_f32 (TRANS ops co-execute with XDL WMMA on CDNA5).
__device__ __forceinline__ float fast_tanh(float x) {
  return 1.0f - 2.0f * fast_rcp(__expf(2.0f * x) + 1.0f);
}
__device__ __forceinline__ void lds_fence() { asm volatile("" ::: "memory"); } // same-wave DS ops are HW-ordered; just stop compiler reordering

// A operand: 16 rows (M) x 32 K, row-major source. Per-lane K runs: {0..7,16..23} lo-half, {8..15,24..31} hi-half.
__device__ __forceinline__ v16bf load_a(const unsigned short* base, int row0, int kbase, int stride) {
  int lane = threadIdx.x & 31, r = lane & 15, hi = lane >> 4;
  const unsigned short* p = base + (row0 + r) * stride + kbase + hi * 8;
  Frag f; f.q[0] = *(const v4u*)p; f.q[1] = *(const v4u*)(p + 16);
  return f.v;
}
// B operand: 16 rows (N) x 32 K (i.e. B^T row-major). Per-lane K runs: {0..7,16..23} lo-half, {8..15,24..31} hi-half.
__device__ __forceinline__ v16bf load_b(const unsigned short* base, int row0, int kbase, int stride) {
  int lane = threadIdx.x & 31, r = lane & 15, hi = lane >> 4;
  const unsigned short* p = base + (row0 + r) * stride + kbase + hi * 8;
  Frag f; f.q[0] = *(const v4u*)p; f.q[1] = *(const v4u*)(p + 16);
  return f.v;
}
// D tile (M=r+hi*8, N=lane&15) stored transposed: buf[N][colbase + hi*8 + r], one b128 per lane.
__device__ __forceinline__ void store_T(unsigned short* buf, int colbase, int stride, const Acc& a) {
  int lane = threadIdx.x & 31, r = lane & 15, hi = lane >> 4;
  v4u p;
  p.x = (unsigned)f2bf(a.f[0]) | ((unsigned)f2bf(a.f[1]) << 16);
  p.y = (unsigned)f2bf(a.f[2]) | ((unsigned)f2bf(a.f[3]) << 16);
  p.z = (unsigned)f2bf(a.f[4]) | ((unsigned)f2bf(a.f[5]) << 16);
  p.w = (unsigned)f2bf(a.f[6]) | ((unsigned)f2bf(a.f[7]) << 16);
  *(v4u*)(buf + r * stride + colbase + hi * 8) = p;
}
__device__ __forceinline__ v8f wmma_bf16(v16bf a, v16bf b, v8f c) {
  return __builtin_amdgcn_wmma_f32_16x16x32_bf16(false, a, false, b, (short)0, c, false, false);
}

__global__ void __launch_bounds__(800, 1)
dymixprop_kernel(const float* __restrict__ x,
                 const float* __restrict__ wl1, const float* __restrict__ bl1g,
                 const float* __restrict__ wl2, const float* __restrict__ bl2g,
                 const float* __restrict__ wm1, const float* __restrict__ bm1g,
                 const float* __restrict__ wm2, const float* __restrict__ bm2g,
                 float* __restrict__ out)
{
  extern __shared__ unsigned char smem[];
  unsigned short* xT  = (unsigned short*)(smem + OFF_XT);   // [v][c] bf16 (B operand / scalar reads)
  unsigned short* x1T = (unsigned short*)(smem + OFF_X1T);  // [v][c]
  unsigned short* x2T = (unsigned short*)(smem + OFF_X2T);
  unsigned short* gA0 = (unsigned short*)(smem + OFF_GA0);  // [c][v] g = h/z  (A operand)
  unsigned short* gA1 = (unsigned short*)(smem + OFF_GA1);
  unsigned short* Eb  = (unsigned short*)(smem + OFF_EB);   // per-wave [16][TS]: E chunk, then hT
  unsigned short* W1b = (unsigned short*)(smem + OFF_W1);
  unsigned short* W2b = (unsigned short*)(smem + OFF_W2);
  unsigned short* Wm1b= (unsigned short*)(smem + OFF_WM1);
  unsigned short* Wm2b= (unsigned short*)(smem + OFF_WM2);
  float* z0s = (float*)(smem + OFF_Z0);
  float* z1s = (float*)(smem + OFF_Z1);
  float* bL1 = (float*)(smem + OFF_BL1);
  float* bL2 = (float*)(smem + OFF_BL2);
  float* bM  = (float*)(smem + OFF_BM);

  const int tid  = threadIdx.x;
  const int wave = tid >> 5;             // 0..24, owns spatial tile `wave`
  const int lane = tid & 31;
  const int r16  = lane & 15, hi = lane >> 4;
  const int n = blockIdx.x / LN, l = blockIdx.x % LN;
  const float* xs = x + (size_t)n * CI * VN * LN + l;   // x[n,c,v,l] at xs[(c*VN+v)*LN]
  const v8f vzero = {0.f,0.f,0.f,0.f,0.f,0.f,0.f,0.f};

  // ---------- phase 1: stage inputs into LDS (bf16) ----------
  for (int i = tid; i < CI * VN; i += 800) {
    int c = i / VN, v = i - c * VN;
    xT[v * TS + c] = f2bf(xs[(size_t)(c * VN + v) * LN]);
  }
  for (int i = tid; i < CI * CI; i += 800) {
    int o = i >> 5, c = i & 31;
    W1b[o * TS + c] = f2bf(wl1[i]);
    W2b[o * TS + c] = f2bf(wl2[i]);
  }
  for (int i = tid; i < CI * 96; i += 800) {
    int o = i / 96, k = i - o * 96;
    Wm1b[o * WS + k] = f2bf(wm1[i]);
    Wm2b[o * WS + k] = f2bf(wm2[i]);
  }
  if (tid < CI) { bL1[tid] = bl1g[tid]; bL2[tid] = bl2g[tid]; bM[tid] = bm1g[tid] + bm2g[tid]; }
  for (int i = tid; i < 416; i += 800) z1s[i] = 0.f;
  for (int i = tid; i < CI * (GS - VN); i += 800) {       // zero K-pad columns of both g buffers
    int c = i / (GS - VN), p = i % (GS - VN);
    gA0[c * GS + VN + p] = 0; gA1[c * GS + VN + p] = 0;
  }
  __syncthreads();

  // ---------- phase 2: x1 = tanh(W1 x + b1), x2 = tanh(W2 x + b2), stored transposed ----------
  {
    v16bf bx = load_b(xT, wave * 16, 0, TS);
    for (int m = 0; m < 2; ++m) {
      const unsigned short* Wb = m ? W2b : W1b;
      const float* bb = m ? bL2 : bL1;
      unsigned short* dstT = (m ? x2T : x1T) + (wave * 16) * TS;
      for (int ct = 0; ct < 2; ++ct) {
        Acc d; d.v = wmma_bf16(load_a(Wb, ct * 16, 0, TS), bx, vzero);
        Acc t;
#pragma unroll
        for (int r = 0; r < 8; ++r) t.f[r] = fast_tanh(d.f[r] + bb[ct * 16 + hi * 8 + r]);
        store_T(dstT, ct * 16, TS, t);
      }
    }
  }
  __syncthreads();

  // ---------- phase 3: softmax denominators (no max needed: |adj|<=32) ----------
  {
    v16bf a1 = load_a(x1T, wave * 16, 0, TS);    // rows v = owned tile
    float rs[8];
#pragma unroll
    for (int r = 0; r < 8; ++r) rs[r] = 0.f;
    for (int j = 0; j < NT; ++j) {
      Acc d; d.v = wmma_bf16(a1, load_b(x2T, j * 16, 0, TS), vzero);  // adj[v,w] tile
      float cs = 0.f;
#pragma unroll
      for (int r = 0; r < 8; ++r) { float e = __expf(d.f[r]); rs[r] += e; cs += e; }
      cs += __shfl_xor(cs, 16);
      if (lane < 16) atomicAdd(&z1s[j * 16 + lane], cs);   // z1[w] = sum_v
    }
#pragma unroll
    for (int m = 1; m <= 8; m <<= 1)
#pragma unroll
      for (int r = 0; r < 8; ++r) rs[r] += __shfl_xor(rs[r], m);
    if (r16 == 0)
#pragma unroll
      for (int r = 0; r < 8; ++r) z0s[wave * 16 + hi * 8 + r] = rs[r]; // z0[v] = sum_w
  }
  __syncthreads();
  for (int i = tid; i < VN; i += 800) { z0s[i] = fast_rcp(z0s[i]); z1s[i] = fast_rcp(z1s[i]); }
  __syncthreads();

  // ---------- phase 4: two diffusion directions, fully fused with the output MLP ----------
  Acc outA[2]; outA[0].v = vzero; outA[1].v = vzero;
  unsigned short* Ew = Eb + wave * (16 * TS);
  unsigned short* gbuf[2] = { gA0, gA1 };

  for (int dir = 0; dir < 2; ++dir) {
    const unsigned short* chT = dir ? x2T : x1T;   // A rows = K chunk
    const unsigned short* owT = dir ? x1T : x2T;   // B rows = owned tile
    const float* rz = dir ? z1s : z0s;
    const unsigned short* Wm = dir ? Wm2b : Wm1b;

    // g0 = x * rz  (h0 = x)
    for (int i = tid; i < CI * VN; i += 800) {
      int c = i / VN, v = i - c * VN;
      gbuf[0][c * GS + v] = f2bf(bf2f(xT[v * TS + c]) * rz[v]);
    }
    __syncthreads();

    // MLP x-term: out += Wm[:,0:32] @ x
    {
      v16bf bx = load_b(xT, wave * 16, 0, TS);
      for (int ct = 0; ct < 2; ++ct)
        outA[ct].v = wmma_bf16(load_a(Wm, ct * 16, 0, WS), bx, outA[ct].v);
    }

    v16bf bOwn = load_b(owT, wave * 16, 0, TS);    // reused for every adj tile this direction
    for (int s = 1; s <= 2; ++s) {
      const unsigned short* gRd = gbuf[(s - 1) & 1];
      unsigned short* gWr = gbuf[s & 1];
      Acc acc[2]; acc[0].v = vzero; acc[1].v = vzero;
      for (int kc = 0; kc < 13; ++kc) {
        int kv = kc * 32;
        // rebuild E chunk [owned 16][K 32] = exp(adj), flash-style (adj never materialized)
        for (int sub = 0; sub < 2; ++sub) {
          int vb = kv + sub * 16;
          Acc e;
          if (vb < VN) {
            Acc d; d.v = wmma_bf16(load_a(chT, vb, 0, TS), bOwn, vzero);
#pragma unroll
            for (int r = 0; r < 8; ++r) e.f[r] = __expf(d.f[r]);
          } else {
#pragma unroll
            for (int r = 0; r < 8; ++r) e.f[r] = 0.f;   // K padding 400..415
          }
          store_T(Ew, sub * 16, TS, e);
        }
        lds_fence();
        v16bf bE = load_b(Ew, 0, 0, TS);
        for (int ct = 0; ct < 2; ++ct)
          acc[ct].v = wmma_bf16(load_a(gRd, ct * 16, kv, GS), bE, acc[ct].v);
      }
      // hnew = alpha*x + (1-alpha)*acc ; store raw -> Ew (as hT), scaled -> gWr
      {
        int vg = wave * 16 + r16;
        float rzv = rz[vg];
        for (int ct = 0; ct < 2; ++ct) {
          Acc hn;
#pragma unroll
          for (int r = 0; r < 8; ++r) {
            int c = ct * 16 + hi * 8 + r;
            hn.f[r] = ALPHA * bf2f(xT[vg * TS + c]) + (1.f - ALPHA) * acc[ct].f[r];
          }
          store_T(Ew, ct * 16, TS, hn);
#pragma unroll
          for (int r = 0; r < 8; ++r) {
            int c = ct * 16 + hi * 8 + r;
            gWr[c * GS + vg] = f2bf(hn.f[r] * rzv);
          }
        }
      }
      lds_fence();
      // MLP h-term: out += Wm[:, 32s:32s+32] @ h   (B = this wave's freshly written hT rows)
      {
        v16bf bh = load_b(Ew, 0, 0, TS);
        for (int ct = 0; ct < 2; ++ct)
          outA[ct].v = wmma_bf16(load_a(Wm, ct * 16, 32 * s, WS), bh, outA[ct].v);
      }
      __syncthreads();   // gWr fully written before next step reads it
    }
  }

  // ---------- epilogue: out[n,o,v,l] = acc + b_mlp1[o] + b_mlp2[o] ----------
  {
    int v = wave * 16 + r16;
#pragma unroll
    for (int ct = 0; ct < 2; ++ct)
#pragma unroll
      for (int r = 0; r < 8; ++r) {
        int o = ct * 16 + hi * 8 + r;
        out[((size_t)(n * 32 + o) * VN + v) * LN + l] = outA[ct].f[r] + bM[o];
      }
  }
}

extern "C" void kernel_launch(void* const* d_in, const int* in_sizes, int n_in,
                              void* d_out, int out_size, void* d_ws, size_t ws_size,
                              hipStream_t stream) {
  (void)in_sizes; (void)n_in; (void)d_ws; (void)ws_size; (void)out_size;
  const float* x   = (const float*)d_in[0];
  const float* wl1 = (const float*)d_in[1];
  const float* bl1 = (const float*)d_in[2];
  const float* wl2 = (const float*)d_in[3];
  const float* bl2 = (const float*)d_in[4];
  const float* wm1 = (const float*)d_in[5];
  const float* bm1 = (const float*)d_in[6];
  const float* wm2 = (const float*)d_in[7];
  const float* bm2 = (const float*)d_in[8];
  dim3 grid(NB * LN), block(800);   // 384 (n,l) slices; 25 wave32s per workgroup
  hipLaunchKernelGGL(dymixprop_kernel, grid, block, SMEM_BYTES, stream,
                     x, wl1, bl1, wl2, bl2, wm1, bm1, wm2, bm2, (float*)d_out);
}
